// ScaledDotProductAttention_27874337751624
// MI455X (gfx1250) — matrix operand order repaired and checked by reading, not compiled, not run on hardware
//
#include <hip/hip_runtime.h>

#define BATCH  4
#define TSEQ   2048
#define DMODEL 1024
#define DHEAD  128

typedef __attribute__((ext_vector_type(16))) __bf16   v16bf;
typedef __attribute__((ext_vector_type(8)))  float    v8f;
typedef __attribute__((ext_vector_type(8)))  unsigned v8u;

__device__ __forceinline__ v16bf as_bf16(v8u u) {
  return __builtin_bit_cast(v16bf, u);
}

// A-matrix (16xK) 16-bit fragment: lane m = lane&15, hi = lane>>4
// VGPR j holds K = 2j + 16*hi, +1  (packed lo/hi 16 bits); contiguous order chosen here, not the ISA table order
__device__ __forceinline__ int kbase_a(int j, int hi) {
  return 2 * j + 16 * hi;
}
// B-matrix (Kx16) 16-bit fragment: lane n = lane&15, hi = lane>>4
// VGPR j holds K = 2j + 16*hi, +1 (same contiguous order as A, not the ISA table order)
__device__ __forceinline__ int kbase_b(int j, int hi) {
  return 2 * j + 16 * hi;
}

__device__ __forceinline__ unsigned pack2bf(float a, float b) {
#if __has_builtin(__builtin_amdgcn_cvt_pk_bf16_f32)
  return __builtin_bit_cast(unsigned, __builtin_amdgcn_cvt_pk_bf16_f32(a, b));
#else
  unsigned short ua = __builtin_bit_cast(unsigned short, (__bf16)a);
  unsigned short ub = __builtin_bit_cast(unsigned short, (__bf16)b);
  return (unsigned)ua | ((unsigned)ub << 16);
#endif
}
__device__ __forceinline__ unsigned short bf16bits(float a) {
  return __builtin_bit_cast(unsigned short, (__bf16)a);
}
__device__ __forceinline__ v8f zero8() {
  v8f z = {0.f, 0.f, 0.f, 0.f, 0.f, 0.f, 0.f, 0.f};
  return z;
}
__device__ __forceinline__ v8f wmma_bf16(v8u a, v8u b, v8f c) {
  return __builtin_amdgcn_wmma_f32_16x16x32_bf16(
      false, as_bf16(a), false, as_bf16(b), (short)0, c, false, false);
}

// ---------------------------------------------------------------------------
// Stage 0: W [DMODEL][DHEAD] f32 -> Wt [DHEAD][DMODEL] bf16 (once; 512 KB)
// ---------------------------------------------------------------------------
__global__ __launch_bounds__(256) void wt_convert(
    const float* __restrict__ W, unsigned short* __restrict__ Wt) {
  const int idx = blockIdx.x * 256 + threadIdx.x;  // over DMODEL*DHEAD
  const int k = idx >> 7;
  const int n = idx & 127;
  Wt[(size_t)n * DMODEL + k] = bf16bits(W[idx]);
}

// ---------------------------------------------------------------------------
// Stage 1: out = X @ W + bias (fp32 X, bf16 Wt). 8 waves/block, 16 rows/wave.
// Epilogue stages the bf16 tile in a per-wave 4 KB LDS slab -> b128 stores.
// transposed==0: out[row][col] (Q, K);  transposed==1: out[b][col][t] (V)
// ---------------------------------------------------------------------------
__global__ __launch_bounds__(256) void proj16_wmma(
    const float* __restrict__ X,            // [B*T, DMODEL]
    const unsigned short* __restrict__ Wt,  // [DHEAD, DMODEL] bf16
    const float* __restrict__ bias,         // [DHEAD]
    unsigned short* __restrict__ out,
    int transposed) {
  __shared__ unsigned short Lt[8 * DHEAD * 16];  // 32 KB; 4 KB per wave
  const int lane = threadIdx.x & 31;
  const int wid = threadIdx.x >> 5;
  unsigned short* Lw = Lt + wid * (DHEAD * 16);
  const int n0 = lane & 15;
  const int hi = lane >> 4;
  const int row0 = (blockIdx.x * 8 + wid) * 16;
  const int arow = row0 + n0;

  v8f acc[8];
#pragma unroll
  for (int nt = 0; nt < 8; ++nt) acc[nt] = zero8();

  for (int kb = 0; kb < DMODEL; kb += 32) {
    v8u A;
#pragma unroll
    for (int j = 0; j < 8; ++j) {
      const int k = kb + kbase_a(j, hi);
      const float2 x = *(const float2*)(X + (size_t)arow * DMODEL + k);
      A[j] = pack2bf(x.x, x.y);
    }
#pragma unroll
    for (int nt = 0; nt < 8; ++nt) {
      const size_t wrow = (size_t)(nt * 16 + n0) * DMODEL + kb;
      v8u Bm;
#pragma unroll
      for (int j = 0; j < 8; ++j) {
        Bm[j] = *(const unsigned*)(Wt + wrow + kbase_b(j, hi));
      }
      acc[nt] = wmma_bf16(A, Bm, acc[nt]);
    }
  }

  const int b = row0 / TSEQ;
  const int trow0 = row0 - b * TSEQ;
  if (transposed) {
    // LDS tile as [n][m] -> b128 segments along t
#pragma unroll
    for (int nt = 0; nt < 8; ++nt) {
      const int n = nt * 16 + n0;
      const float bv = bias[n];
#pragma unroll
      for (int j = 0; j < 8; ++j) {
        Lw[n * 16 + j + 8 * hi] = bf16bits(acc[nt][j] + bv);
      }
    }
    asm volatile("s_wait_dscnt 0x0" ::: "memory");  // per-wave DS in-order
#pragma unroll
    for (int i = 0; i < 8; ++i) {
      const int id = i * 32 + lane;  // 256 chunks of 8 shorts
      const int n = id >> 1;
      const int half = id & 1;
      const uint4 d = *(const uint4*)(Lw + n * 16 + half * 8);
      *(uint4*)(out + ((size_t)b * DHEAD + n) * TSEQ + trow0 + half * 8) = d;
    }
  } else {
    // LDS tile as [m][n] -> b128 segments along n
#pragma unroll
    for (int nt = 0; nt < 8; ++nt) {
      const int n = nt * 16 + n0;
      const float bv = bias[n];
#pragma unroll
      for (int j = 0; j < 8; ++j) {
        Lw[(j + 8 * hi) * DHEAD + n] = bf16bits(acc[nt][j] + bv);
      }
    }
    asm volatile("s_wait_dscnt 0x0" ::: "memory");
#pragma unroll
    for (int i = 0; i < 4; ++i) {
      const int id = i * 32 + lane;  // 128 chunks of 8 shorts
      const int row = id >> 3;
      const int seg = id & 7;
      const uint4 d = *(const uint4*)(Lw + row * DHEAD + seg * 8);
      *(uint4*)(out + ((size_t)row0 + row) * DHEAD + seg * 8) = d;
    }
  }
}

// ---------------------------------------------------------------------------
// Stage 2 step: 32 keys of transposed flash attention.
// S^T = K.Q^T  (lane = query, VGPR = key)  -> per-lane softmax stats.
// O^T += V^T.P^T (lane = query in C/D too) -> lane-local alpha rescale,
// contiguous per-lane output rows. P^T B-fragment: 4x shfl_xor(16) exchange.
// ---------------------------------------------------------------------------
template <bool MASK>
__device__ __forceinline__ void flash_step(
    const unsigned short* __restrict__ Kbb,  // Kb + b*T*DHEAD
    const unsigned short* __restrict__ Vbb,  // Vt + b*DHEAD*T
    int kb, int q_global, int n0, int hi,
    const v8u (&Qf)[4], v8f (&acc)[8], float& mrow, float& lrow) {
  const float scale = 0.08838834764831845f * 1.4426950408889634f;  // 1/sqrt(dk)*log2(e)
  v8f S0 = zero8(), S1 = zero8();
  const size_t krow = (size_t)(kb + n0) * DHEAD;
#pragma unroll
  for (int c = 0; c < 4; ++c) {
    v8u K0, K1;
#pragma unroll
    for (int j = 0; j < 8; ++j) {
      const int k = c * 32 + kbase_a(j, hi);
      K0[j] = *(const unsigned*)(Kbb + krow + k);
      K1[j] = *(const unsigned*)(Kbb + krow + 16 * DHEAD + k);
    }
    S0 = wmma_bf16(K0, Qf[c], S0);
    S1 = wmma_bf16(K1, Qf[c], S1);
  }

  float s0v[8], s1v[8];
  float rmax = -3.0e38f;
#pragma unroll
  for (int j = 0; j < 8; ++j) {
    float a0 = S0[j] * scale;
    float a1 = S1[j] * scale;
    if (MASK) {
      const int key = kb + j + 8 * hi;
      if (key > q_global) a0 = -3.0e38f;
      if (key + 16 > q_global) a1 = -3.0e38f;
    }
    s0v[j] = a0;
    s1v[j] = a1;
    rmax = fmaxf(rmax, fmaxf(a0, a1));
  }
  rmax = fmaxf(rmax, __shfl_xor(rmax, 16, 32));  // merge key halves
  const float mnew = fmaxf(mrow, rmax);
  const float alpha = exp2f(mrow - mnew);

  float psum = 0.0f;
  unsigned pk[8];  // pk[0..3]: keys 8hi..8hi+7 ; pk[4..7]: keys 16+8hi..+7
#pragma unroll
  for (int j = 0; j < 4; ++j) {
    const float e00 = exp2f(s0v[2 * j] - mnew);
    const float e01 = exp2f(s0v[2 * j + 1] - mnew);
    const float e10 = exp2f(s1v[2 * j] - mnew);
    const float e11 = exp2f(s1v[2 * j + 1] - mnew);
    psum += (e00 + e01) + (e10 + e11);
    pk[j] = pack2bf(e00, e01);
    pk[4 + j] = pack2bf(e10, e11);
  }
  psum += __shfl_xor(psum, 16, 32);
  lrow = lrow * alpha + psum;

  // P^T B-fragment: lane needs keys 16hi..16hi+15; trade only the half the
  // partner owns (4 dword shuffles instead of 8)
  unsigned own[4], send[4];
#pragma unroll
  for (int j = 0; j < 4; ++j) {
    own[j]  = hi ? pk[4 + j] : pk[j];
    send[j] = hi ? pk[j] : pk[4 + j];
  }
  unsigned recv[4];
#pragma unroll
  for (int j = 0; j < 4; ++j) {
    recv[j] = (unsigned)__shfl_xor((int)send[j], 16, 32);
  }
  v8u Pf;
#pragma unroll
  for (int j = 0; j < 4; ++j) {
    Pf[j]     = hi ? recv[j] : own[j];
    Pf[4 + j] = hi ? own[j]  : recv[j];
  }

  // lane-local alpha rescale; skip entirely when no lane saw a new max
  if (__ballot(mnew > mrow)) {
#pragma unroll
    for (int nt = 0; nt < 8; ++nt) {
#pragma unroll
      for (int j = 0; j < 8; ++j) acc[nt][j] *= alpha;
    }
  }
  mrow = mnew;

  // O^T tiles: acc[nt] += V^T[nt-tile] . P^T
#pragma unroll
  for (int nt = 0; nt < 8; ++nt) {
    v8u Vf;
    const size_t vbase = (size_t)(nt * 16 + n0) * TSEQ + kb;
#pragma unroll
    for (int j = 0; j < 8; ++j) {
      Vf[j] = *(const unsigned*)(Vbb + vbase + kbase_a(j, hi));
    }
    acc[nt] = wmma_bf16(Vf, Pf, acc[nt]);
  }
}

__global__ __launch_bounds__(256) void flash16_wmma(
    const unsigned short* __restrict__ Qb,  // [B*T, DHEAD] bf16
    const unsigned short* __restrict__ Kb,  // [B*T, DHEAD] bf16
    const unsigned short* __restrict__ Vt,  // [B, DHEAD, T] bf16
    float* __restrict__ Out) {              // [B*T, DHEAD] f32
  const int lane = threadIdx.x & 31;
  const int wid = threadIdx.x >> 5;
  const int n0 = lane & 15;
  const int hi = lane >> 4;
  const int tile = blockIdx.x * 8 + wid;
  const int b = tile >> 7;  // tile / (TSEQ/16)
  const int t = tile & 127;
  const int qr = t * 16;
  const size_t qrow0 = (size_t)b * TSEQ + qr;

  const unsigned short* Kbb = Kb + (size_t)b * TSEQ * DHEAD;
  const unsigned short* Vbb = Vt + (size_t)b * DHEAD * TSEQ;

  // Q as B-fragments (dk contraction), 4 chunks of 32
  v8u Qf[4];
#pragma unroll
  for (int c = 0; c < 4; ++c) {
#pragma unroll
    for (int j = 0; j < 8; ++j) {
      Qf[c][j] = *(const unsigned*)(Qb + (qrow0 + n0) * DHEAD + c * 32 + kbase_b(j, hi));
    }
  }

  v8f acc[8];
#pragma unroll
  for (int nt = 0; nt < 8; ++nt) acc[nt] = zero8();
  float mrow = -3.0e38f;
  float lrow = 0.0f;
  const int q_global = qr + n0;
  const int nblk = t / 2 + 1;  // ceil((qr+16)/32)

  // steady state: no masking possible (kb+31 < qr)
  for (int blk = 0; blk < nblk - 1; ++blk) {
    flash_step<false>(Kbb, Vbb, blk * 32, q_global, n0, hi, Qf, acc, mrow, lrow);
  }
  // peeled last block: causal boundary
  flash_step<true>(Kbb, Vbb, (nblk - 1) * 32, q_global, n0, hi, Qf, acc, mrow, lrow);

  // ---- normalize + store: lane owns query row n0; cols nt*16 + 8hi..+7
  const float linv = 1.0f / lrow;
  float* orow = Out + (qrow0 + n0) * DHEAD + 8 * hi;
#pragma unroll
  for (int nt = 0; nt < 8; ++nt) {
    float4 lo, hi4;
    lo.x = acc[nt][0] * linv; lo.y = acc[nt][1] * linv;
    lo.z = acc[nt][2] * linv; lo.w = acc[nt][3] * linv;
    hi4.x = acc[nt][4] * linv; hi4.y = acc[nt][5] * linv;
    hi4.z = acc[nt][6] * linv; hi4.w = acc[nt][7] * linv;
    *(float4*)(orow + nt * 16) = lo;
    *(float4*)(orow + nt * 16 + 4) = hi4;
  }
}

extern "C" void kernel_launch(void* const* d_in, const int* in_sizes, int n_in,
                              void* d_out, int out_size, void* d_ws, size_t ws_size,
                              hipStream_t stream) {
  const float* X_Q  = (const float*)d_in[0];
  const float* X_KV = (const float*)d_in[1];
  const float* Wq   = (const float*)d_in[2];
  const float* bq   = (const float*)d_in[3];
  const float* Wk   = (const float*)d_in[4];
  const float* bk   = (const float*)d_in[5];
  const float* Wv   = (const float*)d_in[6];
  const float* bv   = (const float*)d_in[7];
  float* Out = (float*)d_out;

  const size_t elems = (size_t)BATCH * TSEQ * DHEAD;   // 1M
  const size_t welems = (size_t)DMODEL * DHEAD;        // 128K
  unsigned short* Qb  = (unsigned short*)d_ws;
  unsigned short* Kbf = Qb + elems;
  unsigned short* Vt  = Kbf + elems;   // [B, DHEAD, T]
  unsigned short* Wtq = Vt + elems;    // [DHEAD, DMODEL]
  unsigned short* Wtk = Wtq + welems;
  unsigned short* Wtv = Wtk + welems;

  const dim3 cgrid(DMODEL * DHEAD / 256);
  wt_convert<<<cgrid, 256, 0, stream>>>(Wq, Wtq);
  wt_convert<<<cgrid, 256, 0, stream>>>(Wk, Wtk);
  wt_convert<<<cgrid, 256, 0, stream>>>(Wv, Wtv);

  const dim3 grid(BATCH * TSEQ / 16 / 8);  // 64 blocks x 8 waves
  proj16_wmma<<<grid, 256, 0, stream>>>(X_Q,  Wtq, bq, Qb,  0);
  proj16_wmma<<<grid, 256, 0, stream>>>(X_KV, Wtk, bk, Kbf, 0);
  proj16_wmma<<<grid, 256, 0, stream>>>(X_KV, Wtv, bv, Vt,  1);
  flash16_wmma<<<grid, 256, 0, stream>>>(Qb, Kbf, Vt, Out);
}
